// MPNN_49598282334748
// MI455X (gfx1250) — compile-verified, matrix operand order checked by reading, not run on hardware
//
#include <hip/hip_runtime.h>

// ---------------------------------------------------------------------------
// MPNN for MI455X (gfx1250, wave32).
// Roofline: per-edge 32x32 weight matrices W_e (= eh @ W2^T, 100000 x 1024)
// would be 205 MB bf16 (> 192 MB L2) -> ~820 MB HBM traffic over T=3 iters if
// materialized.  Instead each iteration recomputes W_e tiles with
// v_wmma_f32_16x16x32_bf16 (26 GFLOP/iter, cheap on the WMMA units) and
// immediately contracts them against out[src] into LDS (ds_add_f32), then
// scatters per-edge messages to aggr.  eh (25.6 MB) + packed W2 (256 KB) stay
// L2-resident across iterations -> compute/L2-bound instead of HBM-bound.
// ---------------------------------------------------------------------------

#define N_NODES  40000
#define N_EDGES  100000
#define N_GRAPHS 1024
#define CC       32
#define F_ATOM   30
#define F_BOND   11
#define H_EDGE   128
#define T_ITER   3

typedef __attribute__((ext_vector_type(16))) __bf16 v16bf;
typedef __attribute__((ext_vector_type(8)))  __bf16 v8bf;
typedef __attribute__((ext_vector_type(8)))  float  v8f;

__device__ __forceinline__ float sigmoidf_(float x) { return 1.0f / (1.0f + expf(-x)); }

__device__ __forceinline__ void atomicMaxF(float* addr, float val) {
    int* ia = (int*)addr;
    int cur = __float_as_int(*addr);
    while (__int_as_float(cur) < val) {
        int old = atomicCAS(ia, cur, __float_as_int(val));
        if (old == cur) break;
        cur = old;
    }
}

// ---------------------------------------------------------------- degree ----
__global__ void deg_accum_kernel(const int* __restrict__ edge_index, float* __restrict__ deg) {
    int e = blockIdx.x * blockDim.x + threadIdx.x;
    if (e < N_EDGES) atomicAdd(&deg[edge_index[N_EDGES + e]], 1.0f);
}
__global__ void deg_invert_kernel(float* __restrict__ deg) {
    int n = blockIdx.x * blockDim.x + threadIdx.x;
    if (n < N_NODES) deg[n] = 1.0f / fmaxf(deg[n], 1.0f);
}

// --------------------------------------------------------------- lin0 -------
__global__ void lin0_relu_kernel(const float* __restrict__ x, const float* __restrict__ W,
                                 const float* __restrict__ b, float* __restrict__ node) {
    int idx = blockIdx.x * blockDim.x + threadIdx.x;   // N_NODES*CC threads
    int c = idx & 31, n = idx >> 5;
    float acc = b[c];
    const float* xr = x + n * F_ATOM;
    const float* wr = W + c * F_ATOM;
    #pragma unroll
    for (int f = 0; f < F_ATOM; ++f) acc += xr[f] * wr[f];
    node[idx] = fmaxf(acc, 0.0f);
}

// ---------------------------------------------------------- edge MLP L1 -----
__global__ void mlp1_relu_kernel(const float* __restrict__ ea, const float* __restrict__ W1,
                                 const float* __restrict__ b1, __bf16* __restrict__ eh) {
    int idx = blockIdx.x * blockDim.x + threadIdx.x;   // N_EDGES*H_EDGE threads
    int hh = idx & (H_EDGE - 1), e = idx >> 7;
    float acc = b1[hh];
    const float* ar = ea + e * F_BOND;
    const float* wr = W1 + hh * F_BOND;
    #pragma unroll
    for (int f = 0; f < F_BOND; ++f) acc += ar[f] * wr[f];
    eh[idx] = (__bf16)fmaxf(acc, 0.0f);
}

// ---- pack W2^T (mlp_W2 is [1024,128]) into WMMA B-fragment layout, bf16 ----
// B element (k, n) = mlp_W2[n*128 + k].  Fragment: lane = nl + 16*(region&1),
// elem = (k%8) + 8*(region>>1), region = (k%32)/8, chunk kc = k/32.
__global__ void pack_w2_kernel(const float* __restrict__ W2, __bf16* __restrict__ pk) {
    int idx = blockIdx.x * blockDim.x + threadIdx.x;   // 1024*128 threads
    int k = idx & 127, n = idx >> 7;
    int nt = n >> 4, nl = n & 15;
    int kc = k >> 5, kl = k & 31;
    int region = kl >> 3, within = kl & 7;
    int lane = nl + 16 * (region & 1);
    int elem = within + 8 * (region >> 1);
    pk[(((size_t)nt * 4 + kc) * 32 + lane) * 16 + elem] = (__bf16)W2[(size_t)n * 128 + k];
}

// ------- FUSED: WMMA recompute of W_e tiles + message contraction -----------
// Per block: 16 edges.  4 waves sweep the 64 n-tiles of the 1024-wide edge
// MLP output.  For tile nt the input channel c = nt>>1 is a wave-uniform
// scalar and o = (nt&1)*16 + nl is lane-distinct, so each lane folds its 8
// accumulator rows into LDS msg[m][o] with ds_add_f32.  out[src] is staged
// TRANSPOSED (sOutT[c][m]) so each lane's 8 operands are 32 contiguous,
// 16B-aligned bytes -> two ds_load_b128 hoisted above the ds_add chain.
__global__ void __launch_bounds__(128)
msg_wmma_kernel(const __bf16* __restrict__ eh, const __bf16* __restrict__ w2pk,
                const float* __restrict__ b2, const float* __restrict__ node,
                const int* __restrict__ edge_index, float* __restrict__ aggr) {
    __shared__ float sOutT[CC][16];      // out[src[e]] transposed: [c][m]
    __shared__ float sMsg[16][CC + 1];   // padded: lanes l and l+16 hit distinct banks

    const int tid  = threadIdx.x;
    const int lane = tid & 31;
    const int wave = tid >> 5;           // 4 waves
    const int e0   = blockIdx.x * 16;
    const int half = lane >> 4;
    const int nl   = lane & 15;

    // stage out[src] (transposed) and zero msg accumulator
    for (int i = tid; i < 16 * CC; i += 128) {
        int m = i >> 5, c = i & 31;      // c fastest -> coalesced global reads
        sOutT[c][m] = node[(size_t)edge_index[e0 + m] * CC + c];
        sMsg[m][c] = 0.0f;
    }
    __syncthreads();

    // A fragments: 16x128 bf16 tile of eh, per CDNA5 16-bit A layout.
    const __bf16* arow = eh + (size_t)(e0 + nl) * H_EDGE;
    v16bf a[4];
    #pragma unroll
    for (int kc = 0; kc < 4; ++kc) {
        int k0 = kc * 32 + half * 8;
        v8bf lo = *(const v8bf*)(arow + k0);        // K = k0   .. k0+7
        v8bf hi = *(const v8bf*)(arow + k0 + 16);   // K = k0+16.. k0+23
        a[kc] = __builtin_shufflevector(lo, hi, 0, 1, 2, 3, 4, 5, 6, 7,
                                                8, 9, 10, 11, 12, 13, 14, 15);
    }

    // Each wave handles 16 of the 64 n-tiles.
    for (int t = 0; t < 16; ++t) {
        const int nt = wave * 16 + t;
        v8f acc = {};
        #pragma unroll
        for (int kc = 0; kc < 4; ++kc) {
            v16bf bfr = *(const v16bf*)(w2pk + (((size_t)nt * 4 + kc) * 32 + lane) * 16);
            acc = __builtin_amdgcn_wmma_f32_16x16x32_bf16(
                      false, a[kc], false, bfr, (short)0, acc, false, false);
        }
        const int c = nt >> 1;                   // input channel: wave-uniform scalar
        const int o = ((nt & 1) << 4) + nl;      // output channel: lane-distinct
        const float bias = b2[c * CC + o];

        // batched LDS read: 8 contiguous floats (two ds_load_b128, one wait)
        float oc[8];
        const float* op = &sOutT[c][8 * half];
        #pragma unroll
        for (int r = 0; r < 8; ++r) oc[r] = op[r];

        #pragma unroll
        for (int r = 0; r < 8; ++r) {
            int m = r + 8 * half;                // edge within tile (C/D layout)
            atomicAdd(&sMsg[m][o], (acc[r] + bias) * oc[r]);   // ds_add_f32
        }
    }
    __syncthreads();

    // scatter per-edge messages to aggr[dst]
    for (int i = tid; i < 16 * CC; i += 128) {
        int m = i >> 5, o = i & 31;
        int dst = edge_index[N_EDGES + e0 + m];
        atomicAdd(&aggr[(size_t)dst * CC + o], sMsg[m][o]);
    }
}

// ------------------- fused NNConv-root + mean + GRU node update -------------
__global__ void node_update_kernel(float* __restrict__ node, const float* __restrict__ aggr,
                                   const float* __restrict__ denominv,
                                   const float* __restrict__ root_W, const float* __restrict__ conv_b,
                                   const float* __restrict__ Wih, const float* __restrict__ Whh,
                                   const float* __restrict__ bih, const float* __restrict__ bhh) {
    __shared__ float sH[8][CC];
    __shared__ float sM[8][CC];
    int tid = threadIdx.x;
    int loc = tid >> 5, c = tid & 31;
    int n = blockIdx.x * 8 + loc;
    float h = node[(size_t)n * CC + c];
    sH[loc][c] = h;
    __syncthreads();
    float root = 0.0f;
    #pragma unroll
    for (int k = 0; k < CC; ++k) root += sH[loc][k] * root_W[k * CC + c];
    float m = fmaxf(aggr[(size_t)n * CC + c] * denominv[n] + root + conv_b[c], 0.0f);
    sM[loc][c] = m;
    __syncthreads();
    float gxr = bih[c], gxz = bih[CC + c], gxn = bih[2 * CC + c];
    float ghr = bhh[c], ghz = bhh[CC + c], ghn = bhh[2 * CC + c];
    #pragma unroll
    for (int k = 0; k < CC; ++k) {
        float mk = sM[loc][k], hk = sH[loc][k];
        gxr += mk * Wih[c * CC + k];
        gxz += mk * Wih[(CC + c) * CC + k];
        gxn += mk * Wih[(2 * CC + c) * CC + k];
        ghr += hk * Whh[c * CC + k];
        ghz += hk * Whh[(CC + c) * CC + k];
        ghn += hk * Whh[(2 * CC + c) * CC + k];
    }
    float r = sigmoidf_(gxr + ghr);
    float z = sigmoidf_(gxz + ghz);
    float nn = tanhf(gxn + r * ghn);
    node[(size_t)n * CC + c] = (1.0f - z) * nn + z * h;
}

// ------------------------------- Set2Set ------------------------------------
__global__ void lstm_step_kernel(float* __restrict__ q_star, float* __restrict__ hs,
                                 float* __restrict__ cs,
                                 const float* __restrict__ Wih, const float* __restrict__ Whh,
                                 const float* __restrict__ bih, const float* __restrict__ bhh) {
    __shared__ float g[4 * CC];
    int b = blockIdx.x, j = threadIdx.x;             // 128 threads
    float acc = bih[j] + bhh[j];
    const float* q = q_star + (size_t)b * (2 * CC);
    const float* hp = hs + (size_t)b * CC;
    #pragma unroll
    for (int k = 0; k < 2 * CC; ++k) acc += q[k] * Wih[j * (2 * CC) + k];
    #pragma unroll
    for (int k = 0; k < CC; ++k) acc += hp[k] * Whh[j * CC + k];
    g[j] = acc;
    __syncthreads();
    if (j < CC) {
        int c = j;
        float ii = sigmoidf_(g[c]);
        float ff = sigmoidf_(g[CC + c]);
        float gg = tanhf(g[2 * CC + c]);
        float oo = sigmoidf_(g[3 * CC + c]);
        float cn = ff * cs[(size_t)b * CC + c] + ii * gg;
        float hn = oo * tanhf(cn);
        cs[(size_t)b * CC + c] = cn;
        hs[(size_t)b * CC + c] = hn;
        q_star[(size_t)b * (2 * CC) + c] = hn;       // first half of q_star
    }
}

__global__ void attn_init_kernel(float* __restrict__ emax, float* __restrict__ esum,
                                 float* __restrict__ q_star) {
    int idx = blockIdx.x * blockDim.x + threadIdx.x; // N_GRAPHS*32 threads
    int c = idx & 31, b = idx >> 5;
    if (c == 0) { emax[b] = -3.4e38f; esum[b] = 0.0f; }
    q_star[(size_t)b * (2 * CC) + CC + c] = 0.0f;    // zero rr half
}

__global__ void attn_dot_kernel(const float* __restrict__ node, const float* __restrict__ hs,
                                const int* __restrict__ batch, float* __restrict__ e_buf,
                                float* __restrict__ emax) {
    int tid = threadIdx.x;
    int n = blockIdx.x * 8 + (tid >> 5);
    int c = tid & 31;
    int b = batch[n];
    float p = node[(size_t)n * CC + c] * hs[(size_t)b * CC + c];
    #pragma unroll
    for (int off = 16; off > 0; off >>= 1) p += __shfl_xor(p, off, 32);
    if (c == 0) { e_buf[n] = p; atomicMaxF(&emax[b], p); }
}

__global__ void attn_exp_kernel(const float* __restrict__ e_buf, const float* __restrict__ emax,
                                const int* __restrict__ batch, float* __restrict__ ee_buf,
                                float* __restrict__ esum) {
    int n = blockIdx.x * blockDim.x + threadIdx.x;
    if (n < N_NODES) {
        int b = batch[n];
        float ee = expf(e_buf[n] - emax[b]);
        ee_buf[n] = ee;
        atomicAdd(&esum[b], ee);
    }
}

__global__ void attn_scatter_kernel(const float* __restrict__ node, const float* __restrict__ ee_buf,
                                    const float* __restrict__ esum, const int* __restrict__ batch,
                                    float* __restrict__ q_star) {
    int tid = threadIdx.x;
    int n = blockIdx.x * 8 + (tid >> 5);
    int c = tid & 31;
    int b = batch[n];
    float w = ee_buf[n] / esum[b];
    atomicAdd(&q_star[(size_t)b * (2 * CC) + CC + c], w * node[(size_t)n * CC + c]);
}

// ------------------------------- readout ------------------------------------
__global__ void readout_kernel(const float* __restrict__ q_star,
                               const float* __restrict__ W1, const float* __restrict__ b1,
                               const float* __restrict__ W2, const float* __restrict__ b2,
                               float* __restrict__ out) {
    int b = blockIdx.x, c = threadIdx.x;             // 32 threads
    float t = b1[c];
    const float* q = q_star + (size_t)b * (2 * CC);
    #pragma unroll
    for (int k = 0; k < 2 * CC; ++k) t += q[k] * W1[c * (2 * CC) + k];
    t = fmaxf(t, 0.0f);
    float p = t * W2[c];
    #pragma unroll
    for (int off = 16; off > 0; off >>= 1) p += __shfl_xor(p, off, 32);
    if (c == 0) out[b] = p + b2[0];
}

// ---------------------------------------------------------------------------
extern "C" void kernel_launch(void* const* d_in, const int* in_sizes, int n_in,
                              void* d_out, int out_size, void* d_ws, size_t ws_size,
                              hipStream_t stream) {
    (void)in_sizes; (void)n_in; (void)out_size; (void)ws_size;
    const float* x         = (const float*)d_in[0];
    const float* edge_attr = (const float*)d_in[1];
    const int*   edge_idx  = (const int*)  d_in[2];
    const int*   batch     = (const int*)  d_in[3];
    const float* lin0_W    = (const float*)d_in[4];
    const float* lin0_b    = (const float*)d_in[5];
    const float* mlp_W1    = (const float*)d_in[6];
    const float* mlp_b1    = (const float*)d_in[7];
    const float* mlp_W2    = (const float*)d_in[8];
    const float* mlp_b2    = (const float*)d_in[9];
    const float* root_W    = (const float*)d_in[10];
    const float* conv_b    = (const float*)d_in[11];
    const float* gru_Wih   = (const float*)d_in[12];
    const float* gru_Whh   = (const float*)d_in[13];
    const float* gru_bih   = (const float*)d_in[14];
    const float* gru_bhh   = (const float*)d_in[15];
    const float* lstm_Wih  = (const float*)d_in[16];
    const float* lstm_Whh  = (const float*)d_in[17];
    const float* lstm_bih  = (const float*)d_in[18];
    const float* lstm_bhh  = (const float*)d_in[19];
    const float* lin1_W    = (const float*)d_in[20];
    const float* lin1_b    = (const float*)d_in[21];
    const float* lin2_W    = (const float*)d_in[22];
    const float* lin2_b    = (const float*)d_in[23];
    float* out = (float*)d_out;

    // -------- workspace carve-up (256B aligned slices) --------
    char* wsb = (char*)d_ws;
    size_t off = 0;
    auto give = [&](size_t bytes) -> char* {
        char* p = wsb + off;
        off += (bytes + 255) & ~(size_t)255;
        return p;
    };
    float*  node     = (float*) give((size_t)N_NODES * CC * 4);
    float*  aggr     = (float*) give((size_t)N_NODES * CC * 4);
    float*  denominv = (float*) give((size_t)N_NODES * 4);
    float*  e_buf    = (float*) give((size_t)N_NODES * 4);
    float*  ee_buf   = (float*) give((size_t)N_NODES * 4);
    float*  hs       = (float*) give((size_t)N_GRAPHS * CC * 4);
    float*  cs       = (float*) give((size_t)N_GRAPHS * CC * 4);
    float*  q_star   = (float*) give((size_t)N_GRAPHS * 2 * CC * 4);
    float*  emax     = (float*) give((size_t)N_GRAPHS * 4);
    float*  esum     = (float*) give((size_t)N_GRAPHS * 4);
    __bf16* eh       = (__bf16*)give((size_t)N_EDGES * H_EDGE * 2);
    __bf16* w2pack   = (__bf16*)give((size_t)1024 * 128 * 2);

    // -------- degree / denominator --------
    hipMemsetAsync(denominv, 0, (size_t)N_NODES * 4, stream);
    deg_accum_kernel<<<(N_EDGES + 255) / 256, 256, 0, stream>>>(edge_idx, denominv);
    deg_invert_kernel<<<(N_NODES + 255) / 256, 256, 0, stream>>>(denominv);

    // -------- lin0 + edge MLP layer 1 + W2 fragment pack (once) --------
    lin0_relu_kernel<<<(N_NODES * CC) / 256, 256, 0, stream>>>(x, lin0_W, lin0_b, node);
    mlp1_relu_kernel<<<(N_EDGES * H_EDGE) / 256, 256, 0, stream>>>(edge_attr, mlp_W1, mlp_b1, eh);
    pack_w2_kernel<<<(1024 * 128) / 256, 256, 0, stream>>>(mlp_W2, w2pack);

    // -------- T message-passing + GRU iterations (fused WMMA message) --------
    for (int t = 0; t < T_ITER; ++t) {
        hipMemsetAsync(aggr, 0, (size_t)N_NODES * CC * 4, stream);
        msg_wmma_kernel<<<N_EDGES / 16, 128, 0, stream>>>(eh, w2pack, mlp_b2, node,
                                                          edge_idx, aggr);
        node_update_kernel<<<N_NODES / 8, 256, 0, stream>>>(node, aggr, denominv, root_W, conv_b,
                                                            gru_Wih, gru_Whh, gru_bih, gru_bhh);
    }

    // -------- Set2Set --------
    hipMemsetAsync(hs, 0, (size_t)N_GRAPHS * CC * 4, stream);
    hipMemsetAsync(cs, 0, (size_t)N_GRAPHS * CC * 4, stream);
    hipMemsetAsync(q_star, 0, (size_t)N_GRAPHS * 2 * CC * 4, stream);
    for (int t = 0; t < T_ITER; ++t) {
        lstm_step_kernel<<<N_GRAPHS, 128, 0, stream>>>(q_star, hs, cs, lstm_Wih, lstm_Whh,
                                                       lstm_bih, lstm_bhh);
        attn_init_kernel<<<(N_GRAPHS * 32) / 256, 256, 0, stream>>>(emax, esum, q_star);
        attn_dot_kernel<<<N_NODES / 8, 256, 0, stream>>>(node, hs, batch, e_buf, emax);
        attn_exp_kernel<<<(N_NODES + 255) / 256, 256, 0, stream>>>(e_buf, emax, batch, ee_buf, esum);
        attn_scatter_kernel<<<N_NODES / 8, 256, 0, stream>>>(node, ee_buf, esum, batch, q_star);
    }

    // -------- readout --------
    readout_kernel<<<N_GRAPHS, 32, 0, stream>>>(q_star, lin1_W, lin1_b, lin2_W, lin2_b, out);
}
